// PointCloudEncoder_23630910062860
// MI455X (gfx1250) — matrix operand order repaired and checked by reading, not compiled, not run on hardware
//
#include <hip/hip_runtime.h>

typedef _Float16 h16;
typedef __attribute__((ext_vector_type(16))) _Float16 v16h;
typedef __attribute__((ext_vector_type(8)))  _Float16 v8h;
typedef __attribute__((ext_vector_type(8)))  float    v8f;

// ---------------------------------------------------------------- transpose in
__global__ void k_transpose_in(const float* __restrict__ in, float* __restrict__ xyz,
                               int B, int N) {
  int i = blockIdx.x * blockDim.x + threadIdx.x;
  if (i >= B * N) return;
  int b = i / N, n = i % N;
#pragma unroll
  for (int c = 0; c < 3; ++c)
    xyz[((size_t)b * N + n) * 3 + c] = in[((size_t)b * 3 + c) * N + n];
}

// ---------------------------------------------------------------- FPS (serial over S, parallel over N)
#define FPS_T 1024
__global__ void __launch_bounds__(FPS_T) k_fps(const float* __restrict__ xyz,
                                               float* __restrict__ dist,
                                               int* __restrict__ outIdx, int N, int S) {
  int b = blockIdx.x;
  const float* X = xyz + (size_t)b * N * 3;
  float* D = dist + (size_t)b * N;
  int* O = outIdx + (size_t)b * S;
  int t = threadIdx.x;
  for (int j = t; j < N; j += FPS_T) D[j] = 1e10f;
  __shared__ float s_val[FPS_T];
  __shared__ int s_idx[FPS_T];
  __shared__ int s_far;
  if (t == 0) s_far = 0;
  __syncthreads();
  for (int s = 0; s < S; ++s) {
    int far = s_far;
    if (t == 0) O[s] = far;
    float cx = X[far * 3 + 0], cy = X[far * 3 + 1], cz = X[far * 3 + 2];
    float bv = -1.f;
    int bi = 0x7fffffff;
    for (int j = t; j < N; j += FPS_T) {
      float dx = X[j * 3 + 0] - cx, dy = X[j * 3 + 1] - cy, dz = X[j * 3 + 2] - cz;
      float d = fminf(D[j], dx * dx + dy * dy + dz * dz);
      D[j] = d;
      if (d > bv) { bv = d; bi = j; }
    }
    s_val[t] = bv; s_idx[t] = bi;
    __syncthreads();
    for (int off = FPS_T / 2; off > 0; off >>= 1) {
      if (t < off) {
        float ov = s_val[t + off]; int oi = s_idx[t + off];
        if (ov > s_val[t] || (ov == s_val[t] && oi < s_idx[t])) { s_val[t] = ov; s_idx[t] = oi; }
      }
      __syncthreads();
    }
    if (t == 0) s_far = s_idx[0];
    __syncthreads();
  }
}

// ---------------------------------------------------------------- gather centers
__global__ void k_gather3(const float* __restrict__ xyz, const int* __restrict__ idx,
                          float* __restrict__ out, int N, int S, int B) {
  int i = blockIdx.x * blockDim.x + threadIdx.x;
  if (i >= B * S) return;
  int b = i / S, s = i % S;
  int j = idx[b * S + s];
#pragma unroll
  for (int c = 0; c < 3; ++c)
    out[(size_t)i * 3 + c] = xyz[((size_t)b * N + j) * 3 + c];
}

// ---------------------------------------------------------------- ball query (first K hits, pad w/ first)
__global__ void k_ballquery(const float* __restrict__ xyz, const float* __restrict__ nx,
                            int* __restrict__ gidx, int N, int S, int K, float r2, int B) {
  int i = blockIdx.x * blockDim.x + threadIdx.x;
  if (i >= B * S) return;
  int b = i / S;
  float cx = nx[(size_t)i * 3 + 0], cy = nx[(size_t)i * 3 + 1], cz = nx[(size_t)i * 3 + 2];
  const float* X = xyz + (size_t)b * N * 3;
  int* G = gidx + (size_t)i * K;
  int cnt = 0, first = 0;
  for (int j = 0; j < N && cnt < K; ++j) {
    float dx = X[j * 3 + 0] - cx, dy = X[j * 3 + 1] - cy, dz = X[j * 3 + 2] - cz;
    if (dx * dx + dy * dy + dz * dz <= r2) {
      if (cnt == 0) first = j;
      G[cnt++] = j;
    }
  }
  for (; cnt < K; ++cnt) G[cnt] = first;
}

// ---------------------------------------------------------------- grouping -> f16 GEMM operand (padded Cin)
__global__ void k_group(const float* __restrict__ xyz, const float* __restrict__ nx,
                        const float* __restrict__ feats, const int* __restrict__ gidx,
                        h16* __restrict__ A, int N, int S, int K, int Cf, int Cpad, int B) {
  size_t M = (size_t)B * S * K;
  size_t m = (size_t)blockIdx.x * blockDim.x + threadIdx.x;
  if (m >= M) return;
  size_t bs = m / K;
  int s = (int)(bs % S), b = (int)(bs / S);
  int j = gidx[m];
  h16* row = A + m * Cpad;
  const float* P = xyz + ((size_t)b * N + j) * 3;
  const float* C = nx + ((size_t)b * S + s) * 3;
  row[0] = (h16)(P[0] - C[0]);
  row[1] = (h16)(P[1] - C[1]);
  row[2] = (h16)(P[2] - C[2]);
  int c = 3;
  if (Cf > 0) {
    const float* F = feats + ((size_t)b * N + j) * Cf;
    for (int q = 0; q < Cf; ++q) row[c++] = (h16)F[q];
  }
  for (; c < Cpad; ++c) row[c] = (h16)0.f;
}

// ---------------------------------------------------------------- pack weights f32[Cout,Cin] -> f16[Cout,Cpad]
__global__ void k_packW(const float* __restrict__ W, h16* __restrict__ Wp,
                        int Cout, int Cin, int Cpad) {
  int i = blockIdx.x * blockDim.x + threadIdx.x;
  if (i >= Cout * Cpad) return;
  int o = i / Cpad, c = i % Cpad;
  Wp[i] = (h16)((c < Cin) ? W[(size_t)o * Cin + c] : 0.f);
}

__global__ void k_zero(float* p, int n) {
  int i = blockIdx.x * blockDim.x + threadIdx.x;
  if (i < n) p[i] = 0.f;
}

// ---------------------------------------------------------------- WMMA GEMM: X = A * Wp^T + bias, f16 out + stats
// grid = (M/64, Cout/64), block = 128 (4 waves). Each wave: 16(M) x 64(N) tile,
// A tile held in registers and reused across 4 B tiles -> 4 back-to-back WMMAs/K-step.
__global__ void __launch_bounds__(128) k_gemm_wmma(const h16* __restrict__ A,
                                                   const h16* __restrict__ Wp,
                                                   const float* __restrict__ bias,
                                                   h16* __restrict__ X,
                                                   float* __restrict__ stats,
                                                   int M, int Cpad, int Cout) {
  int wave = threadIdx.x >> 5;
  int lane = threadIdx.x & 31;
  int m0 = (blockIdx.x * 4 + wave) * 16;
  int n0 = blockIdx.y * 64;
  if (m0 >= M) return;  // wave-uniform; EXEC stays all-ones inside live waves
  int lhalf = lane & 15;
  int hi = lane >> 4;  // 0: lanes 0-15, 1: lanes 16-31
  const h16* Arow = A + (size_t)(m0 + lhalf) * Cpad;
  const h16* Brow0 = Wp + (size_t)(n0 + 0 + lhalf) * Cpad;
  const h16* Brow1 = Wp + (size_t)(n0 + 16 + lhalf) * Cpad;
  const h16* Brow2 = Wp + (size_t)(n0 + 32 + lhalf) * Cpad;
  const h16* Brow3 = Wp + (size_t)(n0 + 48 + lhalf) * Cpad;
  v8f acc0 = {}, acc1 = {}, acc2 = {}, acc3 = {};
  for (int k0 = 0; k0 < Cpad; k0 += 32) {
    if (k0 + 32 < Cpad) __builtin_prefetch(Arow + k0 + 32, 0, 0);  // global_prefetch_b8
    // A tile (16x32 f16), k order chosen here: lane<16 holds K0..K15, lane>=16 K16..K31
    v16h a = *(const v16h*)(Arow + k0 + hi * 16);
    // B tile (32x16 f16), same k order as A: col = lane%16; lane<16 holds K0..K15, lane>=16 K16..K31
    v16h b0 = *(const v16h*)(Brow0 + k0 + hi * 16);
    v16h b1 = *(const v16h*)(Brow1 + k0 + hi * 16);
    v16h b2 = *(const v16h*)(Brow2 + k0 + hi * 16);
    v16h b3 = *(const v16h*)(Brow3 + k0 + hi * 16);
    acc0 = __builtin_amdgcn_wmma_f32_16x16x32_f16(false, a, false, b0, (short)0, acc0, false, false);
    acc1 = __builtin_amdgcn_wmma_f32_16x16x32_f16(false, a, false, b1, (short)0, acc1, false, false);
    acc2 = __builtin_amdgcn_wmma_f32_16x16x32_f16(false, a, false, b2, (short)0, acc2, false, false);
    acc3 = __builtin_amdgcn_wmma_f32_16x16x32_f16(false, a, false, b3, (short)0, acc3, false, false);
  }
  v8f accs[4] = {acc0, acc1, acc2, acc3};
#pragma unroll
  for (int t = 0; t < 4; ++t) {
    int n = n0 + t * 16 + lhalf;
    float bsn = bias ? bias[n] : 0.f;
    float ssum = 0.f, ssq = 0.f;
#pragma unroll
    for (int r = 0; r < 8; ++r) {
      int mrow = m0 + r + hi * 8;  // C/D layout: VGPR r = row M=r (lanes 0-15) / M=8+r (16-31)
      float v = accs[t][r] + bsn;
      X[(size_t)mrow * Cout + n] = (h16)v;
      ssum += v;
      ssq += v * v;
    }
    // fold lanes (n, n+16-partner) then one atomic per channel per wave
    ssum += __shfl_down(ssum, 16, 32);
    ssq  += __shfl_down(ssq, 16, 32);
    if (hi == 0) {
      atomicAdd(&stats[n], ssum);
      atomicAdd(&stats[Cout + n], ssq);
    }
  }
}

// ---------------------------------------------------------------- BN (batch stats) + ReLU, in place
__global__ void k_bn_relu(h16* __restrict__ X, const float* __restrict__ stats,
                          const float* __restrict__ gamma, const float* __restrict__ beta,
                          size_t M, int Cout) {
  size_t i = (size_t)blockIdx.x * blockDim.x + threadIdx.x;
  if (i >= M * (size_t)Cout) return;
  int c = (int)(i % Cout);
  float inv_m = 1.f / (float)M;
  float mean = stats[c] * inv_m;
  float var = stats[Cout + c] * inv_m - mean * mean;
  float inv = rsqrtf(var + 1e-5f);
  float v = ((float)X[i] - mean) * inv * gamma[c] + beta[c];
  X[i] = (h16)fmaxf(v, 0.f);
}

// ---------------------------------------------------------------- max over K
__global__ void k_maxpool(const h16* __restrict__ X, float* __restrict__ out,
                          int S, int K, int C, int B) {
  int i = blockIdx.x * blockDim.x + threadIdx.x;
  if (i >= B * S * C) return;
  int c = i % C;
  size_t bs = (size_t)i / C;
  const h16* P = X + bs * K * (size_t)C + c;
  float m = -3.0e38f;
  for (int k = 0; k < K; ++k) m = fmaxf(m, (float)P[(size_t)k * C]);
  out[i] = m;
}

// ---------------------------------------------------------------- output transposes
__global__ void k_out_xyz(const float* __restrict__ nx, float* __restrict__ out, int S, int B) {
  int i = blockIdx.x * blockDim.x + threadIdx.x;
  if (i >= B * 3 * S) return;
  int b = i / (3 * S), rem = i % (3 * S), c = rem / S, s = rem % S;
  out[i] = nx[((size_t)b * S + s) * 3 + c];
}
__global__ void k_out_feats(const float* __restrict__ f, float* __restrict__ out,
                            int S, int C, int B) {
  int i = blockIdx.x * blockDim.x + threadIdx.x;
  if (i >= B * C * S) return;
  int b = i / (C * S), rem = i % (C * S), c = rem / S, s = rem % S;
  out[i] = f[((size_t)b * S + s) * C + c];
}

// ================================================================ host
extern "C" void kernel_launch(void* const* d_in, const int* in_sizes, int n_in,
                              void* d_out, int out_size, void* d_ws, size_t ws_size,
                              hipStream_t stream) {
  (void)in_sizes; (void)n_in; (void)out_size; (void)ws_size;
  const float* inp = (const float*)d_in[0];

  const int B = 4, N0 = 8192;
  const int Scfg[4] = {2048, 1024, 512, 256};
  const int Kcfg[4] = {64, 32, 16, 16};
  const float Rcfg[4] = {0.2f, 0.4f, 0.8f, 1.2f};
  const int Cin0[4] = {3, 131, 259, 259};
  const int Cpad0[4] = {32, 160, 288, 288};
  const int Outs[4][3] = {{64, 64, 128}, {128, 128, 256}, {128, 128, 256}, {128, 128, 256}};

  char* ws = (char*)d_ws;
  size_t off = 0;
  auto alloc = [&](size_t bytes) -> char* {
    char* p = ws + off;
    off += (bytes + 255) & ~(size_t)255;
    return p;
  };
  float* xyz0 = (float*)alloc((size_t)B * N0 * 3 * 4);
  float* dist = (float*)alloc((size_t)B * N0 * 4);
  int* fpsidx = (int*)alloc((size_t)B * 2048 * 4);
  int* gidx = (int*)alloc((size_t)B * 2048 * 64 * 4);
  float* nx[4];
  for (int i = 0; i < 4; ++i) nx[i] = (float*)alloc((size_t)B * Scfg[i] * 3 * 4);
  float* pooled[2];
  for (int i = 0; i < 2; ++i) pooled[i] = (float*)alloc((size_t)B * 2048 * 256 * 4);
  float* stats = (float*)alloc(512 * 4);
  h16* Wp = (h16*)alloc((size_t)256 * 288 * 2);
  h16* bufA = (h16*)alloc((size_t)34 * 1024 * 1024 * 2);
  h16* bufB = (h16*)alloc((size_t)68 * 1024 * 1024 * 2);

  auto cdiv = [](size_t a, size_t b) -> unsigned { return (unsigned)((a + b - 1) / b); };

  k_transpose_in<<<cdiv((size_t)B * N0, 256), 256, 0, stream>>>(inp, xyz0, B, N0);

  for (int li = 0; li < 4; ++li) {
    const float* xyz_cur = (li == 0) ? xyz0 : nx[li - 1];
    int N = (li == 0) ? N0 : Scfg[li - 1];
    int S = Scfg[li], K = Kcfg[li];
    float r2 = Rcfg[li] * Rcfg[li];
    size_t M = (size_t)B * S * K;

    k_fps<<<B, FPS_T, 0, stream>>>(xyz_cur, dist, fpsidx, N, S);
    k_gather3<<<cdiv((size_t)B * S, 256), 256, 0, stream>>>(xyz_cur, fpsidx, nx[li], N, S, B);
    k_ballquery<<<cdiv((size_t)B * S, 256), 256, 0, stream>>>(xyz_cur, nx[li], gidx, N, S, K, r2, B);

    const float* fprev = (li == 0) ? nullptr : pooled[(li - 1) & 1];
    int Cf = (li == 0) ? 0 : Outs[li - 1][2];
    k_group<<<cdiv(M, 256), 256, 0, stream>>>(xyz_cur, nx[li], fprev, gidx, bufA,
                                              N, S, K, Cf, Cpad0[li], B);

    int cpad_in = Cpad0[li];
    h16* src = bufA;
    h16* dst = bufB;
    for (int mi = 0; mi < 3; ++mi) {
      int cout = Outs[li][mi];
      int cin = (mi == 0) ? Cin0[li] : Outs[li][mi - 1];
      int widx = 1 + (li * 3 + mi) * 4;
      const float* W = (const float*)d_in[widx + 0];
      const float* bias = (const float*)d_in[widx + 1];
      const float* gamma = (const float*)d_in[widx + 2];
      const float* beta = (const float*)d_in[widx + 3];

      k_packW<<<cdiv((size_t)cout * cpad_in, 256), 256, 0, stream>>>(W, Wp, cout, cin, cpad_in);
      k_zero<<<cdiv((size_t)2 * cout, 256), 256, 0, stream>>>(stats, 2 * cout);

      dim3 grid(cdiv(M, 64), (unsigned)(cout / 64));
      k_gemm_wmma<<<grid, 128, 0, stream>>>(src, Wp, bias, dst, stats, (int)M, cpad_in, cout);
      k_bn_relu<<<cdiv(M * (size_t)cout, 256), 256, 0, stream>>>(dst, stats, gamma, beta, M, cout);

      h16* t = src; src = dst; dst = t;  // after mi=2, result sits in bufB (== src)
      cpad_in = cout;
    }
    k_maxpool<<<cdiv((size_t)B * S * Outs[li][2], 256), 256, 0, stream>>>(
        src, pooled[li & 1], S, K, Outs[li][2], B);
  }

  // outputs: xyz^T [B,3,256] then feats^T [B,256,256]
  float* out = (float*)d_out;
  int Sf = Scfg[3], Cfinal = Outs[3][2];
  k_out_xyz<<<cdiv((size_t)B * 3 * Sf, 256), 256, 0, stream>>>(nx[3], out, Sf, B);
  k_out_feats<<<cdiv((size_t)B * Cfinal * Sf, 256), 256, 0, stream>>>(
      pooled[3 & 1], out + (size_t)B * 3 * Sf, Sf, Cfinal, B);
}
